// CNN_quantize_4329327034682
// MI455X (gfx1250) — matrix operand order repaired and checked by reading, not compiled, not run on hardware
//
#include <hip/hip_runtime.h>
#include <hip/hip_bf16.h>
#include <math.h>

typedef __attribute__((ext_vector_type(16))) _Float16 v16h;
typedef __attribute__((ext_vector_type(8)))  _Float16 h8;
typedef __attribute__((ext_vector_type(8)))  float    v8f;

#define G_GROUPS 3
#define K_CODES  128
#define D_DIM    16
#define BSZ      128
#define HID      512
#define OUT_CLS  10
#define MLP_IN   6400          // 16*20*20
#define N_PATCH  221184        // 128*3*96*96/16
#define PTILES   (N_PATCH/16)  // 13824
#define CTILES   (K_CODES/16)  // 8

// ---------------------------------------------------------------------------
// Per-patch / per-code squared norms
// ---------------------------------------------------------------------------
__global__ void k_xnorm(const float* __restrict__ xp, float* __restrict__ xn) {
  int i = blockIdx.x * blockDim.x + threadIdx.x;
  if (i >= N_PATCH) return;
  const float* r = xp + (size_t)i * D_DIM;
  float s = 0.f;
#pragma unroll
  for (int d = 0; d < D_DIM; ++d) s = fmaf(r[d], r[d], s);
  xn[i] = s;
}

__global__ void k_cnorm(const float* __restrict__ cb, float* __restrict__ cn) {
  int i = blockIdx.x * blockDim.x + threadIdx.x;
  if (i >= G_GROUPS * K_CODES) return;
  const float* r = cb + (size_t)i * D_DIM;
  float s = 0.f;
#pragma unroll
  for (int d = 0; d < D_DIM; ++d) s = fmaf(r[d], r[d], s);
  cn[i] = s;
}

// ---------------------------------------------------------------------------
// Fully fused quantize: one wave per 16-patch tile. 3 groups x 8 WMMAs cover
// the whole K=128 code axis in registers; softmax(-d2), argmin, q, mean-over-
// groups output and loss are computed via 16-lane shfl_xor reductions (each
// D row lives in one VGPR index of one half-group). Single pass over att.
// ---------------------------------------------------------------------------
__global__ void k_quant_fused(const float* __restrict__ xp, const float* __restrict__ cb,
                              const float* __restrict__ xnorm, const float* __restrict__ cnorm,
                              float* __restrict__ att, float* __restrict__ xq,
                              float* __restrict__ lossAcc) {
  int wid = (blockIdx.x * blockDim.x + threadIdx.x) >> 5;
  if (wid >= PTILES) return;
  const int pt = wid;
  const int lane = threadIdx.x & 31;
  const int half = lane >> 4;
  const int l16  = lane & 15;

  // A: patch tile, 16-bit A 16x32 layout (upper 16 K zero-padded, D=16)
  v16h a = {};
  {
    const float* row = xp + ((size_t)(pt * 16 + l16)) * D_DIM + half * 8;
#pragma unroll
    for (int j = 0; j < 8; ++j) a[j] = (_Float16)row[j];
  }
  float xn[8];
#pragma unroll
  for (int r = 0; r < 8; ++r) xn[r] = xnorm[pt * 16 + r + 8 * half];

  float outAcc[8];
#pragma unroll
  for (int r = 0; r < 8; ++r) outAcc[r] = 0.f;
  float lossPart = 0.f;

  for (int g = 0; g < G_GROUPS; ++g) {
    v8f acc[CTILES];
#pragma unroll
    for (int t = 0; t < CTILES; ++t) {
      v16h b = {};
      {
        const float* crow = cb + ((size_t)g * K_CODES + t * 16 + l16) * D_DIM + half * 8;
#pragma unroll
        for (int j = 0; j < 8; ++j) b[j] = (_Float16)crow[j];
      }
      v8f c = {};
      acc[t] = __builtin_amdgcn_wmma_f32_16x16x32_f16(false, a, false, b,
                                                      (short)0, c, false, false);
    }
    float cn[CTILES];
#pragma unroll
    for (int t = 0; t < CTILES; ++t) cn[t] = cnorm[g * K_CODES + t * 16 + l16];

#pragma unroll
    for (int r = 0; r < 8; ++r) {
      const size_t patch = (size_t)pt * 16 + r + 8 * half;
      float d2v[CTILES];
#pragma unroll
      for (int t = 0; t < CTILES; ++t) d2v[t] = xn[r] - 2.0f * acc[t][r] + cn[t];
      // argmin over 128 codes (8 local + 16-lane half-group reduce)
      float bv = d2v[0]; int bi = l16;
#pragma unroll
      for (int t = 1; t < CTILES; ++t) {
        int k = t * 16 + l16;
        if (d2v[t] < bv || (d2v[t] == bv && k < bi)) { bv = d2v[t]; bi = k; }
      }
#pragma unroll
      for (int mk = 8; mk; mk >>= 1) {
        float ov = __shfl_xor(bv, mk, 32);
        int   oi = __shfl_xor(bi, mk, 32);
        if (ov < bv || (ov == bv && oi < bi)) { bv = ov; bi = oi; }
      }
      // softmax(-d2) = exp(min_d2 - d2)/sum
      float e[CTILES]; float s = 0.f;
#pragma unroll
      for (int t = 0; t < CTILES; ++t) { e[t] = expf(bv - d2v[t]); s += e[t]; }
#pragma unroll
      for (int mk = 8; mk; mk >>= 1) s += __shfl_xor(s, mk, 32);
      float inv = 1.0f / s;
      float* rowp = att + ((size_t)g * N_PATCH + patch) * K_CODES + l16;
#pragma unroll
      for (int t = 0; t < CTILES; ++t) rowp[t * 16] = e[t] * inv;
      // q accumulate (lane dim = l16, row owned by this half-group)
      float qd  = cb[((size_t)g * K_CODES + bi) * D_DIM + l16];
      float xpd = xp[patch * D_DIM + l16];
      outAcc[r] += qd;
      float df = qd - xpd;
      lossPart = fmaf(df, df, lossPart);
    }
  }
#pragma unroll
  for (int r = 0; r < 8; ++r)
    xq[((size_t)pt * 16 + r + 8 * half) * D_DIM + l16] = outAcc[r] * (1.0f / 3.0f);
#pragma unroll
  for (int mk = 16; mk; mk >>= 1) lossPart += __shfl_xor(lossPart, mk, 32);
  if (lane == 0) atomicAdd(lossAcc, lossPart);
}

__global__ void k_finalize(const float* __restrict__ lossAcc, float* __restrict__ out) {
  out[1280] = lossAcc[0] * (2.0f / (float)((size_t)G_GROUPS * N_PATCH * D_DIM));
  out[1281 + (size_t)G_GROUPS * N_PATCH * K_CODES] = 0.0f;
}

// ---------------------------------------------------------------------------
// Layout transforms (tiny, bandwidth-trivial)
// ---------------------------------------------------------------------------
// xq (NCHW f32) -> x4 (NHWC f16, C padded 3->4 with zeros)
__global__ void k_x4(const float* __restrict__ xq, _Float16* __restrict__ x4) {
  int i = blockIdx.x * blockDim.x + threadIdx.x;
  if (i >= BSZ * 96 * 96 * 4) return;
  int ic = i & 3;
  int s = i >> 2;
  int x = s % 96; int t = s / 96; int y = t % 96; int b = t / 96;
  float v = (ic < 3) ? xq[(((size_t)b * 3 + ic) * 96 + y) * 96 + x] : 0.f;
  x4[i] = (_Float16)v;
}
// w1 (32,3,5,5) -> wt1[oc][k], k = tap*4+ic (pad ic=3 and tap>=25 with 0), K=128
__global__ void k_wt1(const float* __restrict__ w, _Float16* __restrict__ wt) {
  int i = blockIdx.x * blockDim.x + threadIdx.x;
  if (i >= 32 * 128) return;
  int oc = i >> 7, k = i & 127;
  int tap = k >> 2, ic = k & 3;
  float v = (tap < 25 && ic < 3) ? w[((size_t)oc * 3 + ic) * 25 + tap] : 0.f;
  wt[i] = (_Float16)v;
}
// w (OC,32,5,5) -> wt[oc][k], k = tap*32+ic, K=800
__global__ void k_wt_conv(const float* __restrict__ w, _Float16* __restrict__ wt, int OC) {
  int i = blockIdx.x * blockDim.x + threadIdx.x;
  if (i >= OC * 800) return;
  int oc = i / 800, k = i % 800;
  int tap = k >> 5, ic = k & 31;
  wt[i] = (_Float16)w[((size_t)oc * 32 + ic) * 25 + tap];
}
// w_fc1 (512,6400 c-major) -> f16 reordered to NHWC k' = s*16+c
__global__ void k_wfc1(const float* __restrict__ w, _Float16* __restrict__ wt) {
  int i = blockIdx.x * blockDim.x + threadIdx.x;
  if (i >= HID * MLP_IN) return;
  int n = i / MLP_IN, kp = i % MLP_IN;
  int s = kp >> 4, c = kp & 15;
  wt[i] = (_Float16)w[(size_t)n * MLP_IN + c * 400 + s];
}

// ---------------------------------------------------------------------------
// conv1 implicit-GEMM WMMA: x4 NHWC4 f16 -> y1 NHWC f16 (92x92x32), relu.
// M=16 x-pixels, N=16 ocs, K=128 (25 taps x 4 padded ic), 4 WMMAs.
// ---------------------------------------------------------------------------
__global__ void k_conv1_wmma(const _Float16* __restrict__ x4, const _Float16* __restrict__ wt1,
                             const float* __restrict__ bias, _Float16* __restrict__ y1) {
  const int NT = 2, XT = 6, OH = 92;
  int wid = (blockIdx.x * blockDim.x + threadIdx.x) >> 5;
  if (wid >= BSZ * OH * XT * NT) return;
  int nt = wid % NT; int t = wid / NT;
  int xt = t % XT;   t /= XT;
  int oy = t % OH;   int b = t / OH;
  const int lane = threadIdx.x & 31, half = lane >> 4, l16 = lane & 15;
  const int ox = xt * 16 + l16;
  const int oc = nt * 16 + l16;
  const _Float16* brow = wt1 + (size_t)oc * 128;
  v8f acc = {};
#pragma unroll
  for (int kt = 0; kt < 4; ++kt) {
    v16h av, bw;
#pragma unroll
    for (int j = 0; j < 8; ++j) {
      int k0 = kt * 32 + half * 8 + j;
      int k1 = k0 + 16;
      av[j]     = (k0 < 100) ? x4[(((size_t)b * 96 + oy + ((k0 >> 2) / 5)) * 96 + ox + ((k0 >> 2) % 5)) * 4 + (k0 & 3)]
                             : (_Float16)0.f;
      av[8 + j] = (k1 < 100) ? x4[(((size_t)b * 96 + oy + ((k1 >> 2) / 5)) * 96 + ox + ((k1 >> 2) % 5)) * 4 + (k1 & 3)]
                             : (_Float16)0.f;
    }
    const h8 b0 = *(const h8*)(brow + kt * 32 + half * 8);
    const h8 b1 = *(const h8*)(brow + kt * 32 + 16 + half * 8);
#pragma unroll
    for (int j = 0; j < 8; ++j) { bw[j] = b0[j]; bw[8 + j] = b1[j]; }
    acc = __builtin_amdgcn_wmma_f32_16x16x32_f16(false, av, false, bw,
                                                 (short)0, acc, false, false);
  }
  float bn = bias[oc];
#pragma unroll
  for (int r = 0; r < 8; ++r) {
    int xo = xt * 16 + r + 8 * half;
    if (xo < 92) {
      float v = fmaxf(acc[r] + bn, 0.f);
      y1[(((size_t)b * 92 + oy) * 92 + xo) * 32 + oc] = (_Float16)v;
    }
  }
}

// ---------------------------------------------------------------------------
// conv(5x5, IC=32) + 2x2 maxpool + bias + relu, implicit-GEMM WMMA.
// One wave: two adjacent oy rows (sharing B tiles), 16 x-pixels, 16 ocs.
// Pool happens in D registers (x-pairs are adjacent VGPR indices per lane).
// NHWC f16 in -> NHWC f16 pooled out.
// ---------------------------------------------------------------------------
template <int IHW, int OC, int NT, int XT, int PHW>
__global__ void k_convpool_wmma(const _Float16* __restrict__ in, const _Float16* __restrict__ wt,
                                const float* __restrict__ bias, _Float16* __restrict__ out) {
  int wid = (blockIdx.x * blockDim.x + threadIdx.x) >> 5;
  if (wid >= BSZ * PHW * XT * NT) return;
  int nt = wid % NT; int t = wid / NT;
  int xt = t % XT;   t /= XT;
  int py = t % PHW;  int b = t / PHW;
  const int lane = threadIdx.x & 31, half = lane >> 4, l16 = lane & 15;
  const int oy0 = py * 2;
  const int oc  = nt * 16 + l16;
  const _Float16* brow = wt + (size_t)oc * 800;
  v8f acc0 = {}, acc1 = {};
  for (int tap = 0; tap < 25; ++tap) {
    int ky = tap / 5, kx = tap % 5;
    v16h bw;
    {
      const h8 b0 = *(const h8*)(brow + tap * 32 + half * 8);
      const h8 b1 = *(const h8*)(brow + tap * 32 + 16 + half * 8);
#pragma unroll
      for (int j = 0; j < 8; ++j) { bw[j] = b0[j]; bw[8 + j] = b1[j]; }
    }
    const size_t base0 = (((size_t)b * IHW + oy0 + ky) * IHW + xt * 16 + l16 + kx) * 32;
    const size_t base1 = base0 + (size_t)IHW * 32;
    v16h a0, a1;
    {
      const h8 p0 = *(const h8*)(in + base0 + half * 8);
      const h8 p1 = *(const h8*)(in + base0 + 16 + half * 8);
#pragma unroll
      for (int j = 0; j < 8; ++j) { a0[j] = p0[j]; a0[8 + j] = p1[j]; }
    }
    {
      const h8 p0 = *(const h8*)(in + base1 + half * 8);
      const h8 p1 = *(const h8*)(in + base1 + 16 + half * 8);
#pragma unroll
      for (int j = 0; j < 8; ++j) { a1[j] = p0[j]; a1[8 + j] = p1[j]; }
    }
    acc0 = __builtin_amdgcn_wmma_f32_16x16x32_f16(false, a0, false, bw, (short)0, acc0, false, false);
    acc1 = __builtin_amdgcn_wmma_f32_16x16x32_f16(false, a1, false, bw, (short)0, acc1, false, false);
  }
  float bn = bias[oc];
#pragma unroll
  for (int i = 0; i < 4; ++i) {
    int px = xt * 8 + 4 * half + i;
    if (px < PHW) {
      float m0 = fmaxf(fmaxf(acc0[2 * i], acc0[2 * i + 1]),
                       fmaxf(acc1[2 * i], acc1[2 * i + 1]));
      float v = fmaxf(m0 + bn, 0.f);
      out[(((size_t)b * PHW + py) * PHW + px) * OC + oc] = (_Float16)v;
    }
  }
}

// ---------------------------------------------------------------------------
// fc1 WMMA GEMM: out[m,n] = relu(A[m,:].W[n,:] + b[n]); A = p3 (NHWC f16),
// W = reordered f16 weights. M=128, K=6400, N=512.
// ---------------------------------------------------------------------------
__global__ void k_fc1_wmma(const _Float16* __restrict__ A, const _Float16* __restrict__ W,
                           const float* __restrict__ bias, float* __restrict__ out) {
  int wid = (blockIdx.x * blockDim.x + threadIdx.x) >> 5;
  const int NT = HID / 16;
  if (wid >= (BSZ / 16) * NT) return;
  const int nTile = wid % NT;
  const int mTile = wid / NT;
  const int lane = threadIdx.x & 31, half = lane >> 4, l16 = lane & 15;
  const _Float16* arow = A + (size_t)(mTile * 16 + l16) * MLP_IN;
  const _Float16* brow = W + (size_t)(nTile * 16 + l16) * MLP_IN;
  v8f acc = {};
  for (int k0 = 0; k0 < MLP_IN; k0 += 32) {
    v16h a, b;
    const h8 a0 = *(const h8*)(arow + k0 + half * 8);
    const h8 a1 = *(const h8*)(arow + k0 + 16 + half * 8);
    const h8 b0 = *(const h8*)(brow + k0 + half * 8);
    const h8 b1 = *(const h8*)(brow + k0 + 16 + half * 8);
#pragma unroll
    for (int j = 0; j < 8; ++j) { a[j] = a0[j]; a[8 + j] = a1[j]; }
#pragma unroll
    for (int j = 0; j < 8; ++j) { b[j] = b0[j]; b[8 + j] = b1[j]; }
    if (k0 + 32 < MLP_IN) {
      __builtin_prefetch(arow + k0 + 32, 0, 1);
      __builtin_prefetch(brow + k0 + 32, 0, 1);
    }
    acc = __builtin_amdgcn_wmma_f32_16x16x32_f16(false, a, false, b,
                                                 (short)0, acc, false, false);
  }
  const int n = nTile * 16 + l16;
  const float bn = bias[n];
#pragma unroll
  for (int r = 0; r < 8; ++r) {
    int m = mTile * 16 + r + 8 * half;
    out[(size_t)m * HID + n] = fmaxf(acc[r] + bn, 0.f);
  }
}

// fc2: (128,512) @ (10,512)^T + b -> pred
__global__ void k_fc2(const float* __restrict__ h, const float* __restrict__ w,
                      const float* __restrict__ bias, float* __restrict__ pred) {
  int idx = blockIdx.x * blockDim.x + threadIdx.x;
  if (idx >= BSZ * OUT_CLS) return;
  int o = idx % OUT_CLS, m = idx / OUT_CLS;
  const float* hr = h + (size_t)m * HID;
  const float* wr = w + (size_t)o * HID;
  float acc = bias[o];
  for (int k = 0; k < HID; ++k) acc = fmaf(hr[k], wr[k], acc);
  pred[idx] = acc;
}

// ---------------------------------------------------------------------------
extern "C" void kernel_launch(void* const* d_in, const int* in_sizes, int n_in,
                              void* d_out, int out_size, void* d_ws, size_t ws_size,
                              hipStream_t stream) {
  (void)in_sizes; (void)n_in; (void)out_size; (void)ws_size;
  const float* x    = (const float*)d_in[0];
  const float* cb   = (const float*)d_in[1];
  const float* w1   = (const float*)d_in[2];
  const float* b1   = (const float*)d_in[3];
  const float* w2   = (const float*)d_in[4];
  const float* b2   = (const float*)d_in[5];
  const float* w3   = (const float*)d_in[6];
  const float* b3   = (const float*)d_in[7];
  const float* wfc1 = (const float*)d_in[8];
  const float* bfc1 = (const float*)d_in[9];
  const float* wfc2 = (const float*)d_in[10];
  const float* bfc2 = (const float*)d_in[11];

  float* out = (float*)d_out;
  float* att = out + 1281;

  char* ws = (char*)d_ws;
  size_t off = 0;
  auto carve = [&](size_t bytes) -> char* {
    char* p = ws + off;
    off = (off + bytes + 255) & ~(size_t)255;
    return p;
  };
  const size_t PAD = 4096;  // tail pad: edge tiles over-read < 512B
  float*    lossAcc = (float*)carve(sizeof(float));
  float*    xnorm   = (float*)carve((size_t)N_PATCH * 4);
  float*    cnorm   = (float*)carve((size_t)G_GROUPS * K_CODES * 4);
  float*    xq      = (float*)carve((size_t)N_PATCH * D_DIM * 4);
  _Float16* x4      = (_Float16*)carve((size_t)BSZ * 96 * 96 * 4 * 2 + PAD);
  _Float16* y1      = (_Float16*)carve((size_t)BSZ * 92 * 92 * 32 * 2 + PAD);
  _Float16* y2p     = (_Float16*)carve((size_t)BSZ * 44 * 44 * 32 * 2 + PAD);
  _Float16* p3      = (_Float16*)carve((size_t)BSZ * 20 * 20 * 16 * 2 + PAD);
  _Float16* wt1     = (_Float16*)carve((size_t)32 * 128 * 2);
  _Float16* wt2     = (_Float16*)carve((size_t)32 * 800 * 2);
  _Float16* wt3     = (_Float16*)carve((size_t)16 * 800 * 2);
  _Float16* w16     = (_Float16*)carve((size_t)HID * MLP_IN * 2);
  float*    fc1o    = (float*)carve((size_t)BSZ * HID * 4);

  hipMemsetAsync(lossAcc, 0, sizeof(float), stream);

  // --- fused quantize (WMMA + in-register softmax/argmin) ---
  k_xnorm<<<(N_PATCH + 255) / 256, 256, 0, stream>>>(x, xnorm);
  k_cnorm<<<(G_GROUPS * K_CODES + 255) / 256, 256, 0, stream>>>(cb, cnorm);
  k_quant_fused<<<PTILES / 8, 256, 0, stream>>>(x, cb, xnorm, cnorm, att, xq, lossAcc);
  k_finalize<<<1, 1, 0, stream>>>(lossAcc, out);

  // --- layout/weight transforms ---
  k_x4<<<(BSZ * 96 * 96 * 4 + 255) / 256, 256, 0, stream>>>(xq, x4);
  k_wt1<<<(32 * 128 + 255) / 256, 256, 0, stream>>>(w1, wt1);
  k_wt_conv<<<(32 * 800 + 255) / 256, 256, 0, stream>>>(w2, wt2, 32);
  k_wt_conv<<<(16 * 800 + 255) / 256, 256, 0, stream>>>(w3, wt3, 16);
  k_wfc1<<<(HID * MLP_IN + 255) / 256, 256, 0, stream>>>(wfc1, w16);

  // --- WMMA implicit-GEMM CNN ---
  {
    int waves = BSZ * 92 * 6 * 2;   // 141312
    k_conv1_wmma<<<waves / 8, 256, 0, stream>>>(x4, wt1, b1, y1);
  }
  {
    int waves = BSZ * 44 * 6 * 2;   // 67584
    k_convpool_wmma<92, 32, 2, 6, 44><<<waves / 8, 256, 0, stream>>>(y1, wt2, b2, y2p);
  }
  {
    int waves = BSZ * 20 * 3 * 1;   // 7680
    k_convpool_wmma<44, 16, 1, 3, 20><<<waves / 8, 256, 0, stream>>>(y2p, wt3, b3, p3);
  }

  // --- MLP ---
  k_fc1_wmma<<<32, 256, 0, stream>>>(p3, w16, bfc1, fc1o);
  k_fc2<<<(BSZ * OUT_CLS + 255) / 256, 256, 0, stream>>>(fc1o, wfc2, bfc2, out);
}
